// LinearTransformerAgent_90099823936196
// MI455X (gfx1250) — matrix operand order repaired and checked by reading, not compiled, not run on hardware
//
#include <hip/hip_runtime.h>

typedef __attribute__((ext_vector_type(16))) _Float16 v16h;
typedef __attribute__((ext_vector_type(8)))  float    v8f;

constexpr int T = 2048, D = 512, H = 8, DH = 64, L = 4;
constexpr int D_OBS = 64, N_ACTS = 18, DFF = 2048;
constexpr int LDSW = 72;   // LDS row stride in halves (64 + 8 pad)

// branch-free tanh-gelu (v_exp_f32 path)
__device__ __forceinline__ float gelu_tanh(float x) {
    float u = 0.7978845608028654f * (x + 0.044715f * x * x * x);
    float a = fabsf(u);
    float e = __expf(-2.0f * a);
    float th = copysignf((1.0f - e) / (1.0f + e), u);
    return 0.5f * x * (1.0f + th);
}

// ---- CDNA5 async global->LDS copy (16B per lane), tracked by ASYNCcnt ----
__device__ __forceinline__ void async_copy_b128(const _Float16* gsrc, _Float16* lds_dst) {
    unsigned int loff = (unsigned int)(uintptr_t)lds_dst;
    unsigned long long ga = (unsigned long long)(uintptr_t)gsrc;
    asm volatile("global_load_async_to_lds_b128 %0, %1, off"
                 :: "v"(loff), "v"(ga) : "memory");
}
__device__ __forceinline__ void async_wait0() {
    asm volatile("s_wait_asynccnt 0" ::: "memory");
}

// ---- WMMA fragment loaders; contiguous per-lane (=> ds_load_b128 pairs) ----
// A tile in LDS: [64 rows m][LDSW halves], k-slice at koff
__device__ __forceinline__ v16h load_frag_a(const _Float16* sA, int mbase, int koff, int lane) {
    int m  = mbase + (lane & 15);
    int kb = koff + ((lane < 16) ? 0 : 16);
    const _Float16* p = sA + m * LDSW + kb;
    v16h a;
#pragma unroll
    for (int r = 0; r < 8; ++r) a[r] = p[r];
#pragma unroll
    for (int r = 0; r < 8; ++r) a[8 + r] = p[8 + r];
    return a;
}
// B tile in LDS transposed: [64 rows n][LDSW halves], k-slice at koff
__device__ __forceinline__ v16h load_frag_bt(const _Float16* sBt, int nbase, int koff, int lane) {
    int n  = nbase + (lane & 15);
    int kb = koff + ((lane < 16) ? 0 : 16);
    const _Float16* p = sBt + n * LDSW + kb;
    v16h b;
#pragma unroll
    for (int r = 0; r < 16; ++r) b[r] = p[r];
    return b;
}
#define WMMA_F16(A_, B_, C_) \
    __builtin_amdgcn_wmma_f32_16x16x32_f16(false, (A_), false, (B_), (short)0, (C_), false, false)

// stage a 64x64-half tile (rows from src, row stride ldsrc) into LDS via async copies
__device__ __forceinline__ void stage_tile64(const _Float16* src, int ldsrc, int r0, int k0,
                                             _Float16* lds, int tid) {
#pragma unroll
    for (int it = 0; it < 4; ++it) {      // 64 rows x 8 chunks = 512 x 16B
        int task = tid + it * 128;
        int r = task >> 3, c = (task & 7) * 8;
        async_copy_b128(src + (long)(r0 + r) * ldsrc + k0 + c, lds + r * LDSW + c);
    }
}

// ---- serial scan: reset-aware time, cumsum(done), gates ----
__global__ void scan_kernel(const int* __restrict__ done, const int* __restrict__ state_obs,
                            int* __restrict__ timev, int* __restrict__ cv,
                            float* __restrict__ bf, float* __restrict__ af,
                            int* __restrict__ done_any, float* __restrict__ out_time) {
    if (threadIdx.x != 0 || blockIdx.x != 0) return;
    int s0 = state_obs[0];
    int run = 0, cs = 0;
    for (int t = 0; t < T; ++t) {
        int d  = (done[t] != 0) ? 1 : 0;
        int tv = s0 + t;
        int md = tv * d;
        run = (t == 0) ? md : (md > run ? md : run);
        timev[t] = tv - run;
        cs += d;
        cv[t] = cs;
    }
    int clast = cs;
    done_any[0] = (clast > 0) ? 1 : 0;
    for (int t = 0; t < T; ++t) {
        bf[t] = (cv[t] == 0)     ? 1.f : 0.f;
        af[t] = (cv[t] == clast) ? 1.f : 0.f;
    }
    out_time[0] = (float)(timev[T - 1] + 1);
}

// ---- f32 -> f16 with transpose: batch of [R][C] f32 -> batch of [C][R] f16 ----
__global__ void __launch_bounds__(256)
cvt_t_kernel(const float* __restrict__ in, _Float16* __restrict__ outp, int R, int C) {
    __shared__ _Float16 tile[32][33];
    long zb = (long)blockIdx.z * R * C;
    const float* src = in + zb;
    _Float16* dst = outp + zb;
    int r0 = blockIdx.y * 32, c0 = blockIdx.x * 32;
    int lx = threadIdx.x & 31, ly = threadIdx.x >> 5;
#pragma unroll
    for (int p = 0; p < 4; ++p)
        tile[ly + p * 8][lx] = (_Float16)src[(long)(r0 + ly + p * 8) * C + c0 + lx];
    __syncthreads();
#pragma unroll
    for (int p = 0; p < 4; ++p)
        dst[(long)(c0 + ly + p * 8) * R + r0 + lx] = tile[lx][ly + p * 8];
}

// ---- f16 transpose of a [T][512] section (ld=3D) into [512][T], optional gate over t ----
__global__ void __launch_bounds__(256)
transpose_f16_kernel(const _Float16* __restrict__ in, int ldin,
                     _Float16* __restrict__ outp, int ldout,
                     const float* __restrict__ gate) {
    __shared__ _Float16 tile[32][33];
    int t0 = blockIdx.x * 32, c0 = blockIdx.y * 32;
    int lx = threadIdx.x & 31, ly = threadIdx.x >> 5;
#pragma unroll
    for (int p = 0; p < 4; ++p) {
        int t = t0 + ly + p * 8;
        _Float16 v = in[(long)t * ldin + c0 + lx];
        if (gate) v = (_Float16)((float)v * gate[t]);
        tile[ly + p * 8][lx] = v;
    }
    __syncthreads();
#pragma unroll
    for (int p = 0; p < 4; ++p)
        outp[(long)(c0 + ly + p * 8) * ldout + t0 + lx] = tile[lx][ly + p * 8];
}

// ---- token embedding -> X (f32 residual stream) ----
__global__ void __launch_bounds__(128)
embed_kernel(const float* __restrict__ obs, const float* __restrict__ w_obs,
             const float* __restrict__ b_obs, const float* __restrict__ emb_act,
             const int* __restrict__ act_p, const float* __restrict__ rew_p,
             const float* __restrict__ w_rew, const float* __restrict__ b_rew,
             const float* __restrict__ emb_time, const int* __restrict__ timev,
             float* __restrict__ X) {
    __shared__ float so[D_OBS];
    int t = blockIdx.x, tid = threadIdx.x;
    if (tid < D_OBS) so[tid] = obs[t * D_OBS + tid];
    __syncthreads();
    int   act = act_p[t];
    int   tm  = timev[t];
    float rw  = rew_p[t];
    for (int d = tid; d < D; d += 128) {
        float s = 0.f;
#pragma unroll 8
        for (int k = 0; k < D_OBS; ++k) s += so[k] * w_obs[k * D + d];
        s += b_obs[d] + emb_act[act * D + d] + rw * w_rew[d] + b_rew[d] + emb_time[(long)tm * D + d];
        X[(long)t * D + d] = s;
    }
}

// ---- layer norm: f32 in, f16 out ----
__global__ void __launch_bounds__(256)
layernorm_kernel(const float* __restrict__ X, const float* __restrict__ s,
                 const float* __restrict__ b, _Float16* __restrict__ O) {
    __shared__ float red[256];
    int t = blockIdx.x, tid = threadIdx.x;
    const float* xr = X + (long)t * D;
    float ls = 0.f;
    for (int d = tid; d < D; d += 256) ls += xr[d];
    red[tid] = ls; __syncthreads();
    for (int k = 128; k > 0; k >>= 1) { if (tid < k) red[tid] += red[tid + k]; __syncthreads(); }
    float mu = red[0] / (float)D;
    __syncthreads();
    float lq = 0.f;
    for (int d = tid; d < D; d += 256) { float df = xr[d] - mu; lq += df * df; }
    red[tid] = lq; __syncthreads();
    for (int k = 128; k > 0; k >>= 1) { if (tid < k) red[tid] += red[tid + k]; __syncthreads(); }
    float rs = rsqrtf(red[0] / (float)D + 1e-6f);
    for (int d = tid; d < D; d += 256)
        O[(long)t * D + d] = (_Float16)((xr[d] - mu) * rs * s[d] + b[d]);
}

// ---- templated GEMM: C = A @ Bt^T; 64x64 tile, KTILE=64, 4 waves x 8 WMMA/stage
// A: [M][K] row-major (lda); Bt: [N][K] row-major (ldbt). M%64==0, N%64==0, K%64==0.
template <bool BIAS, bool GATE, bool GELU_, bool BETA, bool OUTF, bool STATE>
__global__ void __launch_bounds__(128)
gemm_f16_kernel(const _Float16* __restrict__ A, int lda, long az,
                const _Float16* __restrict__ Bt, int ldbt, long bz,
                float* __restrict__ Cf, _Float16* __restrict__ Ch, int ldc, long cz,
                int M, int N, int K,
                const float* __restrict__ bias, const float* __restrict__ gateRow,
                const float* __restrict__ prev, long pz, const int* __restrict__ DA) {
    __shared__ __align__(16) _Float16 sA[64 * LDSW];
    __shared__ __align__(16) _Float16 sBt[64 * LDSW];
    int tid = threadIdx.x, lane = tid & 31, wid = tid >> 5;
    int z  = blockIdx.z;
    int m0 = blockIdx.y * 64, n0 = blockIdx.x * 64;
    A += (long)z * az; Bt += (long)z * bz;
    if (OUTF) Cf += (long)z * cz; else Ch += (long)z * cz;
    if (STATE) prev += (long)z * pz;
    int wm = (wid >> 1) * 32, wn = (wid & 1) * 32;
    v8f acc[2][2] = {};
    for (int kt = 0; kt < K; kt += 64) {
        stage_tile64(A, lda, m0, kt, sA, tid);
        stage_tile64(Bt, ldbt, n0, kt, sBt, tid);
        async_wait0();
        __syncthreads();
#pragma unroll
        for (int ks = 0; ks < 2; ++ks) {
            v16h fa[2], fb[2];
            fa[0] = load_frag_a(sA, wm, ks * 32, lane);
            fa[1] = load_frag_a(sA, wm + 16, ks * 32, lane);
            fb[0] = load_frag_bt(sBt, wn, ks * 32, lane);
            fb[1] = load_frag_bt(sBt, wn + 16, ks * 32, lane);
#pragma unroll
            for (int ti = 0; ti < 2; ++ti)
#pragma unroll
                for (int tj = 0; tj < 2; ++tj)
                    acc[ti][tj] = WMMA_F16(fa[ti], fb[tj], acc[ti][tj]);
        }
        __syncthreads();
    }
    float prevScale = STATE ? (DA[0] ? 0.f : 1.f) : 0.f;
#pragma unroll
    for (int ti = 0; ti < 2; ++ti)
#pragma unroll
        for (int tj = 0; tj < 2; ++tj) {
            int col   = n0 + wn + tj * 16 + (lane & 15);
            int rbase = m0 + wm + ti * 16 + ((lane < 16) ? 0 : 8);
            long base = (long)rbase * ldc + col;
            float bcol = BIAS ? bias[col] : 0.f;
#pragma unroll
            for (int r = 0; r < 8; ++r) {
                long idx = base + (long)r * ldc;
                float v = acc[ti][tj][r];
                if (GATE) v *= gateRow[rbase + r];
                if (BIAS) v += bcol;
                if (GELU_) v = gelu_tanh(v);
                if (BETA) v += OUTF ? Cf[idx] : (float)Ch[idx];
                if (STATE) v += prevScale * prev[(long)(rbase + r) * DH + col];
                if (OUTF) Cf[idx] = v; else Ch[idx] = (_Float16)v;
            }
        }
}

// ---- scores: S = (Q @ K^T) * causal/segment mask; single KTILE=64 stage ----
__global__ void __launch_bounds__(128)
scores_kernel(const _Float16* __restrict__ Q, int ldq, long qz,
              const _Float16* __restrict__ Km, int ldk, long kz,
              _Float16* __restrict__ S, int ldS, long sz,
              const int* __restrict__ cvec) {
    __shared__ __align__(16) _Float16 sA[64 * LDSW];
    __shared__ __align__(16) _Float16 sBt[64 * LDSW];
    __shared__ int cm[64], cn[64];
    int tid = threadIdx.x, lane = tid & 31, wid = tid >> 5;
    int z  = blockIdx.z;
    int m0 = blockIdx.y * 64, n0 = blockIdx.x * 64;
    Q += (long)z * qz; Km += (long)z * kz; S += (long)z * sz;
    if (tid < 64) cm[tid] = cvec[m0 + tid];
    else          cn[tid - 64] = cvec[n0 + tid - 64];
    int wm = (wid >> 1) * 32, wn = (wid & 1) * 32;
    v8f acc[2][2] = {};
    if (m0 + 63 >= n0) {   // skip fully-masked above-diagonal blocks
        stage_tile64(Q, ldq, m0, 0, sA, tid);
        stage_tile64(Km, ldk, n0, 0, sBt, tid);
        async_wait0();
        __syncthreads();
#pragma unroll
        for (int ks = 0; ks < 2; ++ks) {
            v16h fa[2], fb[2];
            fa[0] = load_frag_a(sA, wm, ks * 32, lane);
            fa[1] = load_frag_a(sA, wm + 16, ks * 32, lane);
            fb[0] = load_frag_bt(sBt, wn, ks * 32, lane);
            fb[1] = load_frag_bt(sBt, wn + 16, ks * 32, lane);
#pragma unroll
            for (int ti = 0; ti < 2; ++ti)
#pragma unroll
                for (int tj = 0; tj < 2; ++tj)
                    acc[ti][tj] = WMMA_F16(fa[ti], fb[tj], acc[ti][tj]);
        }
    } else {
        __syncthreads();   // keep barrier count uniform with staging path
    }
    __syncthreads();       // cm/cn visible to all
#pragma unroll
    for (int ti = 0; ti < 2; ++ti)
#pragma unroll
        for (int tj = 0; tj < 2; ++tj) {
            int lcol   = wn + tj * 16 + (lane & 15);
            int lrbase = wm + ti * 16 + ((lane < 16) ? 0 : 8);
            int cnv = cn[lcol];
            long base = (long)(m0 + lrbase) * ldS + n0 + lcol;
#pragma unroll
            for (int r = 0; r < 8; ++r) {
                int lrow = lrbase + r;
                // branch-free multiplicative mask
                int keep = (int)((m0 + lrow) >= (n0 + lcol)) & (int)(cm[lrow] == cnv);
                float v = acc[ti][tj][r] * (float)keep;
                S[base + (long)r * ldS] = (_Float16)v;
            }
        }
}

// ---- actor head (N=18, VALU path) ----
__global__ void __launch_bounds__(128)
actor_kernel(const _Float16* __restrict__ Hb, const float* __restrict__ w,
             const float* __restrict__ b, float* __restrict__ out) {
    __shared__ float red[128];
    int t = blockIdx.x, tid = threadIdx.x;
    for (int a = 0; a < N_ACTS; ++a) {
        float s = 0.f;
        for (int d = tid; d < D; d += 128)
            s += (float)Hb[(long)t * D + d] * w[(long)d * N_ACTS + a];
        red[tid] = s; __syncthreads();
        for (int k = 64; k > 0; k >>= 1) { if (tid < k) red[tid] += red[tid + k]; __syncthreads(); }
        if (tid == 0) out[(long)t * N_ACTS + a] = red[0] + b[a];
        __syncthreads();
    }
}

// ---- critic head ----
__global__ void __launch_bounds__(128)
critic_kernel(const _Float16* __restrict__ Hb, const float* __restrict__ w,
              const float* __restrict__ b, float* __restrict__ val) {
    __shared__ float red[128];
    int t = blockIdx.x, tid = threadIdx.x;
    float s = 0.f;
    for (int d = tid; d < D; d += 128) s += (float)Hb[(long)t * D + d] * w[d];
    red[tid] = s; __syncthreads();
    for (int k = 64; k > 0; k >>= 1) { if (tid < k) red[tid] += red[tid + k]; __syncthreads(); }
    if (tid == 0) val[t] = red[0] + b[0];
}

extern "C" void kernel_launch(void* const* d_in, const int* in_sizes, int n_in,
                              void* d_out, int out_size, void* d_ws, size_t ws_size,
                              hipStream_t stream) {
    (void)in_sizes; (void)n_in; (void)out_size; (void)ws_size;
    const float* obs        = (const float*)d_in[0];
    const float* rew_p      = (const float*)d_in[1];
    const int*   act_p      = (const int*)d_in[2];
    const int*   done       = (const int*)d_in[3];
    const int*   state_obs  = (const int*)d_in[4];
    const float* state_blks = (const float*)d_in[5];
    const float* w_obs      = (const float*)d_in[6];
    const float* b_obs      = (const float*)d_in[7];
    const float* emb_act    = (const float*)d_in[8];
    const float* w_rew      = (const float*)d_in[9];
    const float* b_rew      = (const float*)d_in[10];
    const float* emb_time   = (const float*)d_in[11];
    const float* ln1_s      = (const float*)d_in[12];
    const float* ln1_b      = (const float*)d_in[13];
    const float* w_qkv      = (const float*)d_in[14];
    const float* b_qkv      = (const float*)d_in[15];
    const float* w_out      = (const float*)d_in[16];
    const float* b_out      = (const float*)d_in[17];
    const float* ln2_s      = (const float*)d_in[18];
    const float* ln2_b      = (const float*)d_in[19];
    const float* w_m1       = (const float*)d_in[20];
    const float* b_m1       = (const float*)d_in[21];
    const float* w_m2       = (const float*)d_in[22];
    const float* b_m2       = (const float*)d_in[23];
    const float* lnf_s      = (const float*)d_in[24];
    const float* lnf_b      = (const float*)d_in[25];
    const float* w_actor    = (const float*)d_in[26];
    const float* b_actor    = (const float*)d_in[27];
    const float* w_critic   = (const float*)d_in[28];
    const float* b_critic   = (const float*)d_in[29];

    // ---- workspace carve-up (256B aligned) ----
    char* p = (char*)d_ws;
    auto alloc = [&](size_t bytes) { void* r = (void*)p; p += (bytes + 255) & ~(size_t)255; return r; };
    float*     X     = (float*)alloc((size_t)T * D * 4);
    _Float16*  Hb16  = (_Float16*)alloc((size_t)T * D * 2);
    _Float16*  QKV16 = (_Float16*)alloc((size_t)T * 3 * D * 2);
    _Float16*  AO16  = (_Float16*)alloc((size_t)T * D * 2);
    _Float16*  MLP16 = (_Float16*)alloc((size_t)T * DFF * 2);
    _Float16*  SC16  = (_Float16*)alloc((size_t)H * T * T * 2);
    _Float16*  KT16  = (_Float16*)alloc((size_t)D * T * 2);
    _Float16*  VT16  = (_Float16*)alloc((size_t)D * T * 2);
    _Float16*  VGT16 = (_Float16*)alloc((size_t)D * T * 2);
    _Float16*  wqT   = (_Float16*)alloc((size_t)L * D * 3 * D * 2);
    _Float16*  woT   = (_Float16*)alloc((size_t)L * D * D * 2);
    _Float16*  w1T   = (_Float16*)alloc((size_t)L * D * DFF * 2);
    _Float16*  w2T   = (_Float16*)alloc((size_t)L * DFF * D * 2);
    _Float16*  sbT   = (_Float16*)alloc((size_t)L * H * DH * DH * 2);
    int*       TIMEV = (int*)alloc(T * 4);
    int*       CV    = (int*)alloc(T * 4);
    float*     BF    = (float*)alloc(T * 4);
    float*     AF    = (float*)alloc(T * 4);
    int*       DA    = (int*)alloc(256);

    float* out        = (float*)d_out;
    float* out_logits = out;
    float* out_val    = out + (size_t)T * N_ACTS;
    float* out_time   = out_val + T;
    float* out_states = out_time + 1;

    scan_kernel<<<1, 1, 0, stream>>>(done, state_obs, TIMEV, CV, BF, AF, DA, out_time);
    cvt_t_kernel<<<dim3(3 * D / 32, D / 32, L), 256, 0, stream>>>(w_qkv, wqT, D, 3 * D);
    cvt_t_kernel<<<dim3(D / 32, D / 32, L), 256, 0, stream>>>(w_out, woT, D, D);
    cvt_t_kernel<<<dim3(DFF / 32, D / 32, L), 256, 0, stream>>>(w_m1, w1T, D, DFF);
    cvt_t_kernel<<<dim3(D / 32, DFF / 32, L), 256, 0, stream>>>(w_m2, w2T, DFF, D);
    cvt_t_kernel<<<dim3(2, 2, L * H), 256, 0, stream>>>(state_blks, sbT, DH, DH);
    embed_kernel<<<T, 128, 0, stream>>>(obs, w_obs, b_obs, emb_act, act_p, rew_p,
                                        w_rew, b_rew, emb_time, TIMEV, X);

    for (int i = 0; i < L; ++i) {
        layernorm_kernel<<<T, 256, 0, stream>>>(X, ln1_s + i * D, ln1_b + i * D, Hb16);

        gemm_f16_kernel<true, false, false, false, false, false>
            <<<dim3(3 * D / 64, T / 64, 1), 128, 0, stream>>>(
            Hb16, D, 0, wqT + (size_t)i * D * 3 * D, D, 0,
            nullptr, QKV16, 3 * D, 0, T, 3 * D, D,
            b_qkv + (size_t)i * 3 * D, nullptr, nullptr, 0, nullptr);

        transpose_f16_kernel<<<dim3(T / 32, D / 32), 256, 0, stream>>>(
            QKV16 + D, 3 * D, KT16, T, nullptr);
        transpose_f16_kernel<<<dim3(T / 32, D / 32), 256, 0, stream>>>(
            QKV16 + 2 * D, 3 * D, VT16, T, nullptr);
        transpose_f16_kernel<<<dim3(T / 32, D / 32), 256, 0, stream>>>(
            QKV16 + 2 * D, 3 * D, VGT16, T, AF);

        scores_kernel<<<dim3(T / 64, T / 64, H), 128, 0, stream>>>(
            QKV16, 3 * D, DH, QKV16 + D, 3 * D, DH, SC16, T, (long)T * T, CV);

        gemm_f16_kernel<false, true, false, false, false, false>
            <<<dim3(1, T / 64, H), 128, 0, stream>>>(
            QKV16, 3 * D, DH, sbT + (size_t)i * H * DH * DH, DH, (long)DH * DH,
            nullptr, AO16, D, DH, T, DH, DH,
            nullptr, BF, nullptr, 0, nullptr);

        gemm_f16_kernel<false, false, false, true, false, false>
            <<<dim3(1, T / 64, H), 128, 0, stream>>>(
            SC16, T, (long)T * T, VT16, T, (long)DH * T,
            nullptr, AO16, D, DH, T, DH, T,
            nullptr, nullptr, nullptr, 0, nullptr);

        gemm_f16_kernel<false, false, false, false, true, true>
            <<<dim3(1, 1, H), 128, 0, stream>>>(
            KT16, T, (long)DH * T, VGT16, T, (long)DH * T,
            out_states + (size_t)i * H * DH * DH, nullptr, DH, (long)DH * DH,
            DH, DH, T, nullptr, nullptr,
            state_blks + (size_t)i * H * DH * DH, (long)DH * DH, DA);

        gemm_f16_kernel<true, false, false, true, true, false>
            <<<dim3(D / 64, T / 64, 1), 128, 0, stream>>>(
            AO16, D, 0, woT + (size_t)i * D * D, D, 0,
            X, nullptr, D, 0, T, D, D,
            b_out + (size_t)i * D, nullptr, nullptr, 0, nullptr);

        layernorm_kernel<<<T, 256, 0, stream>>>(X, ln2_s + i * D, ln2_b + i * D, Hb16);

        gemm_f16_kernel<true, false, true, false, false, false>
            <<<dim3(DFF / 64, T / 64, 1), 128, 0, stream>>>(
            Hb16, D, 0, w1T + (size_t)i * D * DFF, D, 0,
            nullptr, MLP16, DFF, 0, T, DFF, D,
            b_m1 + (size_t)i * DFF, nullptr, nullptr, 0, nullptr);

        gemm_f16_kernel<true, false, false, true, true, false>
            <<<dim3(D / 64, T / 64, 1), 128, 0, stream>>>(
            MLP16, DFF, 0, w2T + (size_t)i * DFF * D, DFF, 0,
            X, nullptr, D, 0, T, D, DFF,
            b_m2 + (size_t)i * D, nullptr, nullptr, 0, nullptr);
    }

    layernorm_kernel<<<T, 256, 0, stream>>>(X, lnf_s, lnf_b, Hb16);
    actor_kernel<<<T, 128, 0, stream>>>(Hb16, w_actor, b_actor, out_logits);
    critic_kernel<<<T, 128, 0, stream>>>(Hb16, w_critic, b_critic, out_val);
}
